// Prealign_68616397521532
// MI455X (gfx1250) — compile-verified, matrix operand order checked by reading, not run on hardware
//
#include <hip/hip_runtime.h>

typedef __attribute__((ext_vector_type(16))) _Float16 v16h;
typedef __attribute__((ext_vector_type(8)))  float    v8f;

// Problem constants
constexpr int NB  = 16;    // batch
constexpr int LD  = 4096;  // doc length
constexpr int LQ  = 128;   // query length
constexpr int E   = 768;   // embed dim (K of projection)
constexpr int H   = 768;   // hidden dim
constexpr int KT_E = E / 32;   // 24 k-tiles of 32 over E/H reduction
constexpr int NT_H = H / 16;   // 48 n-tiles of 16 over H
constexpr int NT_Q = LQ / 16;  // 8  n-tiles over Lq
constexpr int KT_Q = LQ / 32;  // 4  k-tiles over Lq
constexpr int BLK  = 512;      // halves per packed 32x16 B-block (1 KB)

// ---------------- fragment helpers ----------------

__device__ __forceinline__ v8f wmma_f16(v16h a, v16h b, v8f c) {
  return __builtin_amdgcn_wmma_f32_16x16x32_f16(false, a, false, b,
                                                (short)0, c, false, false);
}

// A fragment (16x32 f16) from row-major f16 buffer (LDS), stride in halves.
__device__ __forceinline__ v16h load_a_f16(const _Float16* base, int strideH,
                                           int kbase, int lane) {
  int m = lane & 15, half = lane >> 4;
  const _Float16* row = base + m * strideH;
  union { uint4 u[2]; v16h h; } f;
  f.u[0] = *reinterpret_cast<const uint4*>(row + kbase + half * 8);
  f.u[1] = *reinterpret_cast<const uint4*>(row + kbase + 16 + half * 8);
  return f.h;
}

// A fragment built from row-major f32 global memory (convert to f16 in regs)
__device__ __forceinline__ v16h load_a_f32g(const float* base, int strideF,
                                            int kbase, int lane) {
  int m = lane & 15, half = lane >> 4;
  const float* row = base + (size_t)m * strideF;
  float4 x0 = *reinterpret_cast<const float4*>(row + kbase + half * 8);
  float4 x1 = *reinterpret_cast<const float4*>(row + kbase + half * 8 + 4);
  float4 x2 = *reinterpret_cast<const float4*>(row + kbase + 16 + half * 8);
  float4 x3 = *reinterpret_cast<const float4*>(row + kbase + 16 + half * 8 + 4);
  v16h h;
  h[0]=(_Float16)x0.x; h[1]=(_Float16)x0.y; h[2]=(_Float16)x0.z; h[3]=(_Float16)x0.w;
  h[4]=(_Float16)x1.x; h[5]=(_Float16)x1.y; h[6]=(_Float16)x1.z; h[7]=(_Float16)x1.w;
  h[8]=(_Float16)x2.x; h[9]=(_Float16)x2.y; h[10]=(_Float16)x2.z; h[11]=(_Float16)x2.w;
  h[12]=(_Float16)x3.x; h[13]=(_Float16)x3.y; h[14]=(_Float16)x3.z; h[15]=(_Float16)x3.w;
  return h;
}

// B fragment (32x16 f16) from pre-packed block array: block = 512 halves,
// lane's 16 halves contiguous at blockBase + lane*16  -> two b128 loads.
__device__ __forceinline__ v16h load_b_pack(const _Float16* pack, int blk,
                                            int lane) {
  const _Float16* p = pack + (size_t)blk * BLK + lane * 16;
  union { uint4 u[2]; v16h h; } f;
  f.u[0] = *reinterpret_cast<const uint4*>(p);
  f.u[1] = *reinterpret_cast<const uint4*>(p + 8);
  return f.h;
}

// map (lane, j) of a B/A fragment to its K row inside a 32-row tile
__device__ __forceinline__ int frag_k(int lane, int j) {
  int half = lane >> 4;
  return (j < 8) ? (half * 8 + j) : (16 + half * 8 + (j - 8));
}

// ---------------- kernel 0: pack W f32 -> f16 B-fragment blocks ----------------

__global__ void k_pack_w(const float* __restrict__ W, _Float16* __restrict__ Wp) {
  int g = blockIdx.x * blockDim.x + threadIdx.x;
  if (g >= KT_E * NT_H * BLK) return;
  int blk  = g >> 9;          // (kt*NT_H + nt)
  int e    = g & 511;
  int lane = e >> 4;
  int j    = e & 15;
  int kt = blk / NT_H, nt = blk % NT_H;
  int n = nt * 16 + (lane & 15);
  int k = kt * 32 + frag_k(lane, j);
  Wp[g] = (_Float16)W[(size_t)k * H + n];
}

// ---------------- kernel 1: query_o = relu(Q @ W), emit packed B copies ------

__global__ void k_query(const float* __restrict__ Q,
                        const _Float16* __restrict__ Wp,
                        _Float16* __restrict__ qoT,   // B for score GEMM (K=h)
                        _Float16* __restrict__ qoP) { // B for out GEMM  (K=q)
  __shared__ _Float16 sQo[32 * H];  // 48 KB
  int b  = blockIdx.x >> 2;
  int qt = blockIdx.x & 3;          // which 32-row q tile
  int q0 = qt * 32;
  int tid = threadIdx.x, lane = tid & 31, w = tid >> 5;  // 8 waves
  int mt = w & 1;            // which 16-row half
  int ntbase = w >> 1;       // 0..3

  const float* arow = Q + (size_t)(b * LQ + q0 + mt * 16) * E;

  // register-tiled + software-pipelined: 12 accs, prefetch next A/B fragment
  v8f acc[12];
#pragma unroll
  for (int i = 0; i < 12; ++i) acc[i] = (v8f){};

  v16h aCur = load_a_f32g(arow, E, 0, lane);
  v16h bNext = load_b_pack(Wp, ntbase, lane);
  for (int kt = 0; kt < KT_E; ++kt) {
    v16h aNext = aCur;
#pragma unroll
    for (int i = 0; i < 12; ++i) {
      v16h bCur = bNext;
      if (i < 11) {
        bNext = load_b_pack(Wp, kt * NT_H + ntbase + 4 * (i + 1), lane);
      } else if (kt < KT_E - 1) {
        aNext = load_a_f32g(arow, E, (kt + 1) * 32, lane);
        bNext = load_b_pack(Wp, (kt + 1) * NT_H + ntbase, lane);
      }
      acc[i] = wmma_f16(aCur, bCur, acc[i]);
    }
    aCur = aNext;
  }
#pragma unroll
  for (int i = 0; i < 12; ++i) {
    int nt = ntbase + 4 * i;
    int n = nt * 16 + (lane & 15), half = lane >> 4;
#pragma unroll
    for (int r = 0; r < 8; ++r) {
      float v = fmaxf(acc[i][r], 0.0f);
      sQo[(mt * 16 + r + 8 * half) * H + n] = (_Float16)v;
    }
  }
  __syncthreads();

  // qoP: K = q (this WG covers exactly k-tile qt); blocks (qt*NT_H + nt)
  {
    _Float16* dst = qoP + (size_t)b * (KT_Q * NT_H * BLK)
                        + (size_t)qt * NT_H * BLK;
    for (int idx = tid; idx < NT_H * BLK; idx += 256) {
      int nt = idx >> 9, e = idx & 511, l2 = e >> 4, j = e & 15;
      int n  = nt * 16 + (l2 & 15);
      int kq = frag_k(l2, j);               // local q row 0..31
      dst[idx] = sQo[kq * H + n];
    }
  }
  // qoT: K = h, N = q; this WG covers q n-tiles {q0/16, q0/16+1}
  {
    _Float16* dstT = qoT + (size_t)b * (KT_E * NT_Q * BLK);
    int ntq0 = q0 / 16;
    for (int idx = tid; idx < KT_E * 2 * BLK; idx += 256) {
      int blk = idx >> 9;               // 0..47
      int kth = blk >> 1;
      int ntq = ntq0 + (blk & 1);
      int e = idx & 511, l2 = e >> 4, j = e & 15;
      int qloc = (ntq * 16 + (l2 & 15)) - q0;     // 0..31
      int h    = kth * 32 + frag_k(l2, j);
      dstT[(size_t)(kth * NT_Q + ntq) * BLK + e] = sQo[qloc * H + h];
    }
  }
}

// ---------------- kernel 2: fused doc-proj -> score -> softmax -> out --------

__global__ void k_main(const float* __restrict__ D,
                       const _Float16* __restrict__ Wp,
                       const _Float16* __restrict__ qoT,
                       const _Float16* __restrict__ qoP,
                       float* __restrict__ out) {
  __shared__ _Float16 sDoc[16 * H];     // 24 KB  f16 doc tile
  __shared__ _Float16 sDocO[16 * H];    // 24 KB  relu(doc@W) f16
  __shared__ float    sScore[16 * LQ];  // 8 KB
  __shared__ _Float16 sAttn[16 * LQ];   // 4 KB

  int b  = blockIdx.x / (LD / 16);
  int m0 = (blockIdx.x % (LD / 16)) * 16;
  int tid = threadIdx.x, lane = tid & 31, w = tid >> 5;  // 4 waves

  // stage doc tile (16 contiguous rows x 768) f32 -> f16 LDS, coalesced b128
  const float4* src = reinterpret_cast<const float4*>(D + (size_t)(b * LD + m0) * E);
  for (int i = tid; i < 16 * E / 4; i += 128) {
    float4 v = src[i];
    int o = i * 4;
    sDoc[o + 0] = (_Float16)v.x; sDoc[o + 1] = (_Float16)v.y;
    sDoc[o + 2] = (_Float16)v.z; sDoc[o + 3] = (_Float16)v.w;
  }
  __syncthreads();

  // Phase A: doc_o = relu(doc @ W), 12 accs, 1-deep prefetch of A/B frags
  {
    v8f acc[12];
#pragma unroll
    for (int i = 0; i < 12; ++i) acc[i] = (v8f){};
    v16h aCur = load_a_f16(sDoc, H, 0, lane);
    v16h bNext = load_b_pack(Wp, w, lane);
    for (int kt = 0; kt < KT_E; ++kt) {
      v16h aNext = aCur;
#pragma unroll
      for (int i = 0; i < 12; ++i) {
        v16h bCur = bNext;
        if (i < 11) {
          bNext = load_b_pack(Wp, kt * NT_H + w + 4 * (i + 1), lane);
        } else if (kt < KT_E - 1) {
          aNext = load_a_f16(sDoc, H, (kt + 1) * 32, lane);
          bNext = load_b_pack(Wp, (kt + 1) * NT_H + w, lane);
        }
        acc[i] = wmma_f16(aCur, bCur, acc[i]);
      }
      aCur = aNext;
    }
#pragma unroll
    for (int i = 0; i < 12; ++i) {
      int nt = w + 4 * i;
      int n = nt * 16 + (lane & 15), half = lane >> 4;
#pragma unroll
      for (int r = 0; r < 8; ++r) {
        float v = fmaxf(acc[i][r], 0.0f);
        sDocO[(r + 8 * half) * H + n] = (_Float16)v;
      }
    }
  }
  __syncthreads();

  // Phase B: score = doc_o @ query_o^T (2 q-tiles per wave), pipelined
  {
    const _Float16* qoTb = qoT + (size_t)b * (KT_E * NT_Q * BLK);
    v8f acc0 = {}, acc1 = {};
    int ntq0 = w * 2, ntq1 = w * 2 + 1;
    v16h a  = load_a_f16(sDocO, H, 0, lane);
    v16h b0 = load_b_pack(qoTb, ntq0, lane);
    v16h b1 = load_b_pack(qoTb, ntq1, lane);
    for (int kt = 0; kt < KT_E; ++kt) {
      v16h aN = a, b0N = b0, b1N = b1;
      if (kt < KT_E - 1) {
        aN  = load_a_f16(sDocO, H, (kt + 1) * 32, lane);
        b0N = load_b_pack(qoTb, (kt + 1) * NT_Q + ntq0, lane);
        b1N = load_b_pack(qoTb, (kt + 1) * NT_Q + ntq1, lane);
      }
      acc0 = wmma_f16(a, b0, acc0);
      acc1 = wmma_f16(a, b1, acc1);
      a = aN; b0 = b0N; b1 = b1N;
    }
    int half = lane >> 4;
    int n0 = ntq0 * 16 + (lane & 15);
    int n1 = ntq1 * 16 + (lane & 15);
#pragma unroll
    for (int r = 0; r < 8; ++r) {
      sScore[(r + 8 * half) * LQ + n0] = acc0[r];
      sScore[(r + 8 * half) * LQ + n1] = acc1[r];
    }
  }
  __syncthreads();

  // Phase C: softmax over q (rows of 128), one row per thread 0..15
  if (tid < 16) {
    const float* row = sScore + tid * LQ;
    float m = -3.4e38f;
    for (int q = 0; q < LQ; ++q) m = fmaxf(m, row[q]);
    float s = 0.0f;
    for (int q = 0; q < LQ; ++q) s += __expf(row[q] - m);
    float inv = 1.0f / s;
    for (int q = 0; q < LQ; ++q)
      sAttn[tid * LQ + q] = (_Float16)(__expf(row[q] - m) * inv);
  }
  __syncthreads();

  // Phase D: out = attn @ query_o, 12 accs, K=128 (4 kt), pipelined
  {
    const _Float16* qoPb = qoP + (size_t)b * (KT_Q * NT_H * BLK);
    float* outb = out + (size_t)(b * LD + m0) * H;
    v8f acc[12];
#pragma unroll
    for (int i = 0; i < 12; ++i) acc[i] = (v8f){};
    v16h aCur = load_a_f16(sAttn, LQ, 0, lane);
    v16h bNext = load_b_pack(qoPb, w, lane);
#pragma unroll
    for (int kt = 0; kt < KT_Q; ++kt) {
      v16h aNext = aCur;
#pragma unroll
      for (int i = 0; i < 12; ++i) {
        v16h bCur = bNext;
        if (i < 11) {
          bNext = load_b_pack(qoPb, kt * NT_H + w + 4 * (i + 1), lane);
        } else if (kt < KT_Q - 1) {
          aNext = load_a_f16(sAttn, LQ, (kt + 1) * 32, lane);
          bNext = load_b_pack(qoPb, (kt + 1) * NT_H + w, lane);
        }
        acc[i] = wmma_f16(aCur, bCur, acc[i]);
      }
      aCur = aNext;
    }
#pragma unroll
    for (int i = 0; i < 12; ++i) {
      int nt = w + 4 * i;
      int n = nt * 16 + (lane & 15), half = lane >> 4;
#pragma unroll
      for (int r = 0; r < 8; ++r)
        outb[(size_t)(r + 8 * half) * H + n] = acc[i][r];
    }
  }
}

// ---------------- host-side launcher ----------------

extern "C" void kernel_launch(void* const* d_in, const int* in_sizes, int n_in,
                              void* d_out, int out_size, void* d_ws, size_t ws_size,
                              hipStream_t stream) {
  const float* doc   = (const float*)d_in[0];  // [16,4096,768] f32
  const float* query = (const float*)d_in[1];  // [16,128,768]  f32
  const float* W     = (const float*)d_in[2];  // [768,768]     f32
  float* out = (float*)d_out;                  // [16,4096,768] f32

  // workspace layout (bytes)
  const size_t wpBytes  = (size_t)KT_E * NT_H * BLK * 2;            // 1,179,648
  const size_t qoTBytes = (size_t)NB * KT_E * NT_Q * BLK * 2;       // 3,145,728
  _Float16* Wp  = (_Float16*)d_ws;
  _Float16* qoT = (_Float16*)((char*)d_ws + wpBytes);
  _Float16* qoP = (_Float16*)((char*)d_ws + wpBytes + qoTBytes);

  int wElems = KT_E * NT_H * BLK;
  k_pack_w<<<(wElems + 255) / 256, 256, 0, stream>>>(W, Wp);
  k_query<<<NB * (LQ / 32), 256, 0, stream>>>(query, Wp, qoT, qoP);
  k_main<<<NB * (LD / 16), 128, 0, stream>>>(doc, Wp, qoT, qoP, out);
}
